// HSMSSD_32246614459065
// MI455X (gfx1250) — matrix operand order repaired and checked by reading, not compiled, not run on hardware
//
#include <hip/hip_runtime.h>

#define BQ 16
#define DM 384
#define LQ 4096
#define HQ 64
#define WQ 64
#define SQ 64
#define CDQ 192
#define LDK 40          // padded LDS row stride (elems): 80B avoids 4-way bank conflicts
#define KCH 4           // split-K chunks for k4
#define KLEN (LQ / KCH) // 1024

typedef __attribute__((ext_vector_type(16))) __bf16 v16bf;
typedef __attribute__((ext_vector_type(4)))  __bf16 v4bf;
typedef __attribute__((ext_vector_type(8)))  float  v8f;

// ---- WMMA helpers: bf16 16x16x32, f32 accumulate ---------------------------
// A-matrix 16x32 (MxK) row-major LDS [m][LDK]: lane m = lane&15, half = lane>>4;
// two contiguous 8-elem runs at k = 8*half and k = 16+8*half -> 2x ds_load_b128.
__device__ __forceinline__ v16bf frag_a(const __bf16* lds, int row0, int lane) {
  const __bf16* p = lds + (row0 + (lane & 15)) * LDK + ((lane >> 4) << 3);
  v16bf a;
#pragma unroll
  for (int e = 0; e < 8; ++e) a[e] = p[e];
#pragma unroll
  for (int e = 0; e < 8; ++e) a[8 + e] = p[16 + e];
  return a;
}
// B-matrix 32x16 (KxN) stored TRANSPOSED in LDS as [n][LDK]: lane n = lane&15,
// two contiguous 8-elem runs at k = 8*half and k = 16+8*half -> 2x ds_load_b128.
__device__ __forceinline__ v16bf frag_b_t(const __bf16* lds, int col0, int lane) {
  const __bf16* p = lds + (col0 + (lane & 15)) * LDK + ((lane >> 4) << 3);
  v16bf bb;
#pragma unroll
  for (int e = 0; e < 8; ++e) bb[e] = p[e];
#pragma unroll
  for (int e = 0; e < 8; ++e) bb[8 + e] = p[16 + e];
  return bb;
}
__device__ __forceinline__ v8f wmma_bf16(v16bf a, v16bf b, v8f c) {
  return __builtin_amdgcn_wmma_f32_16x16x32_bf16(false, a, false, b, (short)0, c,
                                                 false, false);
}
__device__ __forceinline__ void cvt4(__bf16* dst, const float4 f) {
  v4bf o;
  o[0] = (__bf16)f.x; o[1] = (__bf16)f.y; o[2] = (__bf16)f.z; o[3] = (__bf16)f.w;
  *reinterpret_cast<v4bf*>(dst) = o;
}
// C/D: VGPR r holds M=row0+r (lanes 0-15) / row0+r+8 (lanes 16-31), N = col0+(lane&15)

// ---- K1: BCdt = BCdt_w (192x384) @ x (384xL) + bias, double-buffered -------
__global__ __launch_bounds__(256) void k1_bcdt(const float* __restrict__ x,
                                               const float* __restrict__ Wm,
                                               const float* __restrict__ bias,
                                               float* __restrict__ out) {
  __shared__ alignas(16) __bf16 sW[2][CDQ * LDK];
  __shared__ alignas(16) __bf16 sX[2][64 * LDK];   // B transposed [n][k]
  int b = blockIdx.y, l0 = blockIdx.x * 64;
  int tid = threadIdx.x, lane = tid & 31, wave = tid >> 5;
  v8f zf = {};
  v8f acc[6];
#pragma unroll
  for (int i = 0; i < 6; ++i) acc[i] = zf;
  const float* xb = x + (size_t)b * DM * LQ;

  auto stage = [&](int buf, int kk) {
    for (int i = tid; i < 1536; i += 256) {         // W: contiguous k both sides
      int r = i >> 3, k4 = (i & 7) << 2;
      cvt4(&sW[buf][r * LDK + k4],
           *reinterpret_cast<const float4*>(&Wm[r * DM + kk + k4]));
    }
    for (int i = tid; i < 512; i += 256) {          // x: coalesced n, scatter to [n][k]
      int k = i >> 4, n4 = (i & 15) << 2;
      const float4 f =
          *reinterpret_cast<const float4*>(&xb[(size_t)(kk + k) * LQ + l0 + n4]);
      sX[buf][(n4 + 0) * LDK + k] = (__bf16)f.x;
      sX[buf][(n4 + 1) * LDK + k] = (__bf16)f.y;
      sX[buf][(n4 + 2) * LDK + k] = (__bf16)f.z;
      sX[buf][(n4 + 3) * LDK + k] = (__bf16)f.w;
    }
  };
  stage(0, 0);
  __syncthreads();
  for (int kk = 0; kk < DM; kk += 32) {
    int cur = (kk >> 5) & 1;
    if (kk + 32 < DM) stage(cur ^ 1, kk + 32);
#pragma unroll
    for (int i = 0; i < 6; ++i) {                   // 48 tiles (12m x 4n) / 8 waves
      int t = wave + 8 * i, mt = t >> 2, nt = t & 3;
      acc[i] = wmma_bf16(frag_a(sW[cur], mt * 16, lane),
                         frag_b_t(sX[cur], nt * 16, lane), acc[i]);
    }
    __syncthreads();
  }
  int n = lane & 15, hm = (lane >> 4) * 8;
#pragma unroll
  for (int i = 0; i < 6; ++i) {
    int t = wave + 8 * i, mt = t >> 2, nt = t & 3;
#pragma unroll
    for (int r = 0; r < 8; ++r) {
      int row = mt * 16 + hm + r;
      out[((size_t)b * CDQ + row) * LQ + l0 + nt * 16 + n] = acc[i][r] + bias[row];
    }
  }
}

// ---- K2: depthwise 3x3, one (b,ch) per block; halo is pure zero padding ----
__global__ __launch_bounds__(256) void k2_dwconv(const float* __restrict__ in,
                                                 const float* __restrict__ w,
                                                 const float* __restrict__ bias,
                                                 float* __restrict__ out) {
  __shared__ float tile[66 * 66];
  int ch = blockIdx.x, b = blockIdx.y, tid = threadIdx.x;
  const float* src = in + ((size_t)b * CDQ + ch) * LQ;
  for (int i = tid; i < 66; i += 256) { tile[i] = 0.f; tile[65 * 66 + i] = 0.f; }
  for (int i = tid; i < 64; i += 256) {
    tile[(i + 1) * 66] = 0.f;
    tile[(i + 1) * 66 + 65] = 0.f;
  }
  for (int i = tid; i < 1024; i += 256) {           // interior: float4 loads
    int r = i >> 4, c4 = (i & 15) << 2;
    const float4 f = *reinterpret_cast<const float4*>(&src[r * 64 + c4]);
    float* d = &tile[(r + 1) * 66 + c4 + 1];
    d[0] = f.x; d[1] = f.y; d[2] = f.z; d[3] = f.w;
  }
  float wr[9];
#pragma unroll
  for (int i = 0; i < 9; ++i) wr[i] = w[ch * 9 + i];
  float bv = bias[ch];
  __syncthreads();
  float* dst = out + ((size_t)b * CDQ + ch) * LQ;
  for (int p = tid; p < LQ; p += 256) {
    int hh = p >> 6, ww = p & 63;
    const float* t0 = &tile[hh * 66 + ww];
    float s = bv;
#pragma unroll
    for (int ky = 0; ky < 3; ++ky)
#pragma unroll
      for (int kx = 0; kx < 3; ++kx) s += wr[ky * 3 + kx] * t0[ky * 66 + kx];
    dst[p] = s;
  }
}

// ---- K3: AB = softmax_L(dt + A[s]) * Bm, one (b,s) per block ---------------
__global__ __launch_bounds__(256) void k3_softmax_ab(const float* __restrict__ conv,
                                                     const float* __restrict__ A,
                                                     float* __restrict__ AB) {
  int s = blockIdx.x, b = blockIdx.y;
  int tid = threadIdx.x, lane = tid & 31, wave = tid >> 5;
  const float* dt = conv + ((size_t)b * CDQ + 2 * SQ + s) * LQ;
  const float* Bm = conv + ((size_t)b * CDQ + s) * LQ;
  float As = A[s];
  __shared__ float red[8];
  float m = -3.0e38f;
  for (int l = tid * 4; l < LQ; l += 1024) {
    const float4 f = *reinterpret_cast<const float4*>(&dt[l]);
    m = fmaxf(m, fmaxf(fmaxf(f.x, f.y), fmaxf(f.z, f.w)));
  }
  m += As;  // constant shift along L
#pragma unroll
  for (int off = 16; off > 0; off >>= 1) m = fmaxf(m, __shfl_xor(m, off));
  if (lane == 0) red[wave] = m;
  __syncthreads();
  if (tid == 0) {
    float mm = red[0];
#pragma unroll
    for (int i = 1; i < 8; ++i) mm = fmaxf(mm, red[i]);
    red[0] = mm;
  }
  __syncthreads();
  m = red[0];
  __syncthreads();
  float ssum = 0.f;
  for (int l = tid * 4; l < LQ; l += 1024) {
    const float4 f = *reinterpret_cast<const float4*>(&dt[l]);
    ssum += __expf(f.x + As - m) + __expf(f.y + As - m) +
            __expf(f.z + As - m) + __expf(f.w + As - m);
  }
#pragma unroll
  for (int off = 16; off > 0; off >>= 1) ssum += __shfl_xor(ssum, off);
  if (lane == 0) red[wave] = ssum;
  __syncthreads();
  if (tid == 0) {
    float t = 0.f;
#pragma unroll
    for (int i = 0; i < 8; ++i) t += red[i];
    red[0] = t;
  }
  __syncthreads();
  float inv = 1.f / red[0];
  float* ab = AB + ((size_t)b * SQ + s) * LQ;
  for (int l = tid * 4; l < LQ; l += 1024) {
    const float4 f = *reinterpret_cast<const float4*>(&dt[l]);
    const float4 g = *reinterpret_cast<const float4*>(&Bm[l]);
    float4 o;
    o.x = __expf(f.x + As - m) * inv * g.x;
    o.y = __expf(f.y + As - m) * inv * g.y;
    o.z = __expf(f.z + As - m) * inv * g.z;
    o.w = __expf(f.w + As - m) * inv * g.w;
    *reinterpret_cast<float4*>(&ab[l]) = o;
  }
}

// ---- K4: h = x (384xL) @ AB^T (Lx64), split-K x4, double-buffered ----------
__global__ __launch_bounds__(256) void k4_h(const float* __restrict__ x,
                                            const float* __restrict__ AB,
                                            float* __restrict__ part) {
  __shared__ alignas(16) __bf16 sA[2][64 * LDK];
  __shared__ alignas(16) __bf16 sB[2][64 * LDK];   // B transposed [n=s][k=l]
  int b = blockIdx.z, chunk = blockIdx.y, m0 = blockIdx.x * 64;
  int base = chunk * KLEN;
  int tid = threadIdx.x, lane = tid & 31, wave = tid >> 5;
  v8f zf = {};
  v8f acc[2];
  acc[0] = zf; acc[1] = zf;
  const float* xb = x + (size_t)b * DM * LQ;
  const float* abb = AB + (size_t)b * SQ * LQ;

  auto stage = [&](int buf, int kk) {
    for (int i = tid; i < 512; i += 256) {          // A: contiguous k both sides
      int r = i >> 3, k4 = (i & 7) << 2;
      cvt4(&sA[buf][r * LDK + k4],
           *reinterpret_cast<const float4*>(&xb[(size_t)(m0 + r) * LQ + kk + k4]));
    }
    for (int i = tid; i < 512; i += 256) {          // B: AB is (s,l) -> contiguous k
      int n = i >> 3, k4 = (i & 7) << 2;
      cvt4(&sB[buf][n * LDK + k4],
           *reinterpret_cast<const float4*>(&abb[(size_t)n * LQ + kk + k4]));
    }
  };
  stage(0, base);
  __syncthreads();
  for (int kk = base; kk < base + KLEN; kk += 32) {
    int cur = ((kk - base) >> 5) & 1;
    if (kk + 32 < base + KLEN) stage(cur ^ 1, kk + 32);
#pragma unroll
    for (int i = 0; i < 2; ++i) {                   // 16 tiles (4x4) / 8 waves
      int t = wave + 8 * i, mt = t >> 2, nt = t & 3;
      acc[i] = wmma_bf16(frag_a(sA[cur], mt * 16, lane),
                         frag_b_t(sB[cur], nt * 16, lane), acc[i]);
    }
    __syncthreads();
  }
  int n = lane & 15, hm = (lane >> 4) * 8;
  float* hb = part + ((size_t)chunk * BQ + b) * DM * SQ;
#pragma unroll
  for (int i = 0; i < 2; ++i) {
    int t = wave + 8 * i, mt = t >> 2, nt = t & 3;
#pragma unroll
    for (int r = 0; r < 8; ++r)
      hb[(size_t)(m0 + mt * 16 + hm + r) * SQ + nt * 16 + n] = acc[i][r];
  }
}

// ---- K4b: deterministic split-K reduction ----------------------------------
__global__ __launch_bounds__(256) void k4_reduce(const float* __restrict__ part,
                                                 float* __restrict__ h) {
  const int per = BQ * DM * SQ;
  int i4 = (blockIdx.x * 256 + threadIdx.x) * 4;
  if (i4 >= per) return;
  float4 s = *reinterpret_cast<const float4*>(&part[i4]);
#pragma unroll
  for (int c = 1; c < KCH; ++c) {
    const float4 f = *reinterpret_cast<const float4*>(&part[(size_t)c * per + i4]);
    s.x += f.x; s.y += f.y; s.z += f.z; s.w += f.w;
  }
  *reinterpret_cast<float4*>(&h[i4]) = s;
}

// ---- Generic small GEMM: out(b,Mtot,64) = W(Mtot x384) @ in(b,384,64) + bias
__global__ __launch_bounds__(256) void k_gemm_ws(const float* __restrict__ Wm,
                                                 const float* __restrict__ bias,
                                                 const float* __restrict__ in,
                                                 float* __restrict__ out, int Mtot) {
  __shared__ alignas(16) __bf16 sW[128 * LDK];
  __shared__ alignas(16) __bf16 sH[64 * LDK];      // B transposed [n][k]
  int b = blockIdx.y, m0 = blockIdx.x * 128;
  int tid = threadIdx.x, lane = tid & 31, wave = tid >> 5;
  v8f zf = {};
  v8f acc[4];
#pragma unroll
  for (int i = 0; i < 4; ++i) acc[i] = zf;
  const float* ib = in + (size_t)b * DM * SQ;
  for (int kk = 0; kk < DM; kk += 32) {
    for (int i = tid; i < 1024; i += 256) {
      int r = i >> 3, k4 = (i & 7) << 2;
      cvt4(&sW[r * LDK + k4],
           *reinterpret_cast<const float4*>(&Wm[(size_t)(m0 + r) * DM + kk + k4]));
    }
    for (int i = tid; i < 512; i += 256) {          // in: coalesced n, scatter [n][k]
      int k = i >> 4, n4 = (i & 15) << 2;
      const float4 f =
          *reinterpret_cast<const float4*>(&ib[(size_t)(kk + k) * SQ + n4]);
      sH[(n4 + 0) * LDK + k] = (__bf16)f.x;
      sH[(n4 + 1) * LDK + k] = (__bf16)f.y;
      sH[(n4 + 2) * LDK + k] = (__bf16)f.z;
      sH[(n4 + 3) * LDK + k] = (__bf16)f.w;
    }
    __syncthreads();
#pragma unroll
    for (int i = 0; i < 4; ++i) {                   // 32 tiles (8x4) / 8 waves
      int t = wave + 8 * i, mt = t >> 2, nt = t & 3;
      acc[i] = wmma_bf16(frag_a(sW, mt * 16, lane),
                         frag_b_t(sH, nt * 16, lane), acc[i]);
    }
    __syncthreads();
  }
  int n = lane & 15, hm = (lane >> 4) * 8;
#pragma unroll
  for (int i = 0; i < 4; ++i) {
    int t = wave + 8 * i, mt = t >> 2, nt = t & 3;
#pragma unroll
    for (int r = 0; r < 8; ++r) {
      int row = m0 + mt * 16 + hm + r;
      out[((size_t)b * Mtot + row) * SQ + nt * 16 + n] = acc[i][r] + bias[row];
    }
  }
}

// ---- K5b: h2 = h1*silu(z) + h1*D, float4 streamed --------------------------
__global__ __launch_bounds__(256) void k5_gate(const float* __restrict__ hz,
                                               const float* __restrict__ Dv,
                                               float* __restrict__ h2) {
  const int total = BQ * DM * SQ;
  int i4 = (blockIdx.x * 256 + threadIdx.x) * 4;
  if (i4 >= total) return;
  int b = i4 / (DM * SQ);
  int rem = i4 - b * (DM * SQ);
  const float* p1 = hz + (size_t)b * 2 * DM * SQ + rem;
  const float4 a = *reinterpret_cast<const float4*>(p1);
  const float4 z = *reinterpret_cast<const float4*>(p1 + DM * SQ);
  float dv = Dv[0];
  float4 o;
  o.x = a.x * (z.x / (1.f + __expf(-z.x))) + a.x * dv;
  o.y = a.y * (z.y / (1.f + __expf(-z.y))) + a.y * dv;
  o.z = a.z * (z.z / (1.f + __expf(-z.z))) + a.z * dv;
  o.w = a.w * (z.w / (1.f + __expf(-z.w))) + a.w * dv;
  *reinterpret_cast<float4*>(&h2[i4]) = o;
}

// ---- K6: y = hout (384x64) @ C (64xL) per batch ----------------------------
__global__ __launch_bounds__(256) void k6_y(const float* __restrict__ hout,
                                            const float* __restrict__ conv,
                                            float* __restrict__ y) {
  __shared__ alignas(16) __bf16 sA[192 * LDK];
  __shared__ alignas(16) __bf16 sB[64 * LDK];      // B transposed [n][k]
  int b = blockIdx.z, m0 = blockIdx.y * 192, l0 = blockIdx.x * 64;
  int tid = threadIdx.x, lane = tid & 31, wave = tid >> 5;
  v8f zf = {};
  v8f acc[6];
#pragma unroll
  for (int i = 0; i < 6; ++i) acc[i] = zf;
  const float* hb = hout + (size_t)b * DM * SQ;
  const float* Cb = conv + ((size_t)b * CDQ + SQ) * LQ;   // C = channels 64..127
  for (int kk = 0; kk < SQ; kk += 32) {
    for (int i = tid; i < 1536; i += 256) {
      int r = i >> 3, k4 = (i & 7) << 2;
      cvt4(&sA[r * LDK + k4],
           *reinterpret_cast<const float4*>(&hb[(size_t)(m0 + r) * SQ + kk + k4]));
    }
    for (int i = tid; i < 512; i += 256) {          // C: coalesced n, scatter [n][k]
      int k = i >> 4, n4 = (i & 15) << 2;
      const float4 f =
          *reinterpret_cast<const float4*>(&Cb[(size_t)(kk + k) * LQ + l0 + n4]);
      sB[(n4 + 0) * LDK + k] = (__bf16)f.x;
      sB[(n4 + 1) * LDK + k] = (__bf16)f.y;
      sB[(n4 + 2) * LDK + k] = (__bf16)f.z;
      sB[(n4 + 3) * LDK + k] = (__bf16)f.w;
    }
    __syncthreads();
#pragma unroll
    for (int i = 0; i < 6; ++i) {                   // 48 tiles (12x4) / 8 waves
      int t = wave + 8 * i, mt = t >> 2, nt = t & 3;
      acc[i] = wmma_bf16(frag_a(sA, mt * 16, lane),
                         frag_b_t(sB, nt * 16, lane), acc[i]);
    }
    __syncthreads();
  }
  int n = lane & 15, hm = (lane >> 4) * 8;
#pragma unroll
  for (int i = 0; i < 6; ++i) {
    int t = wave + 8 * i, mt = t >> 2, nt = t & 3;
#pragma unroll
    for (int r = 0; r < 8; ++r)
      y[((size_t)b * DM + m0 + mt * 16 + hm + r) * LQ + l0 + nt * 16 + n] = acc[i][r];
  }
}

extern "C" void kernel_launch(void* const* d_in, const int* in_sizes, int n_in,
                              void* d_out, int out_size, void* d_ws, size_t ws_size,
                              hipStream_t stream) {
  const float* x    = (const float*)d_in[0];
  // d_in[1]=H, d_in[2]=W (compile-time 64)
  const float* BCw  = (const float*)d_in[3];
  const float* BCb  = (const float*)d_in[4];
  const float* dww  = (const float*)d_in[5];
  const float* dwb  = (const float*)d_in[6];
  const float* hzw  = (const float*)d_in[7];
  const float* hzb  = (const float*)d_in[8];
  const float* outw = (const float*)d_in[9];
  const float* outb = (const float*)d_in[10];
  const float* Av   = (const float*)d_in[11];
  const float* Dv   = (const float*)d_in[12];

  float* ws        = (float*)d_ws;
  float* bcdt_pre  = ws;
  float* bcdt_conv = bcdt_pre + (size_t)BQ * CDQ * LQ;
  float* AB        = bcdt_conv + (size_t)BQ * CDQ * LQ;
  float* h_part    = AB + (size_t)BQ * SQ * LQ;
  float* h         = h_part + (size_t)KCH * BQ * DM * SQ;
  float* hz        = h + (size_t)BQ * DM * SQ;
  float* h2        = hz + (size_t)BQ * 2 * DM * SQ;

  float* y    = (float*)d_out;
  float* hout = y + (size_t)BQ * DM * LQ;  // second tuple output

  k1_bcdt<<<dim3(LQ / 64, BQ), 256, 0, stream>>>(x, BCw, BCb, bcdt_pre);
  k2_dwconv<<<dim3(CDQ, BQ), 256, 0, stream>>>(bcdt_pre, dww, dwb, bcdt_conv);
  k3_softmax_ab<<<dim3(SQ, BQ), 256, 0, stream>>>(bcdt_conv, Av, AB);
  k4_h<<<dim3(DM / 64, KCH, BQ), 256, 0, stream>>>(x, AB, h_part);
  k4_reduce<<<dim3(BQ * DM * SQ / 4 / 256), 256, 0, stream>>>(h_part, h);
  k_gemm_ws<<<dim3(6, BQ), 256, 0, stream>>>(hzw, hzb, h, hz, 2 * DM);
  k5_gate<<<dim3((BQ * DM * SQ / 4 + 255) / 256), 256, 0, stream>>>(hz, Dv, h2);
  k_gemm_ws<<<dim3(3, BQ), 256, 0, stream>>>(outw, outb, h2, hout, DM);
  k6_y<<<dim3(LQ / 64, 2, BQ), 256, 0, stream>>>(hout, bcdt_conv, y);
}
